// OptimizedThermodynamicDetector_13005160972782
// MI455X (gfx1250) — matrix operand order repaired and checked by reading, not compiled, not run on hardware
//
#include <hip/hip_runtime.h>
#include <hip/hip_bf16.h>

// ---------------- problem constants ----------------
#define NN 8192
#define DD 256
#define EPS 1e-8f
#define SPLITS 8              // j-dimension splits (grid.y)
#define RB 256                // rows per block
#define JC 32                 // j columns per chunk
#define JRANGE (NN / SPLITS)  // 1024
#define CHUNKS (JRANGE / JC)  // 32
#define LDW 33                // padded LDS row stride (floats) for G / adj tiles
#define EJS 264               // padded LDS row stride (bf16 elems) for embJ

// ---------------- vector types ----------------
typedef __attribute__((ext_vector_type(16))) __bf16 v16bf;
typedef __attribute__((ext_vector_type(8)))  float  v8f;
typedef __attribute__((ext_vector_type(4)))  unsigned int v4u;
typedef __attribute__((ext_vector_type(8)))  int    v8i;
typedef __attribute__((ext_vector_type(4)))  int    v4i;

union ABu { v16bf v; uint4 q[2]; };

#if defined(__AMDGCN__) && __has_builtin(__builtin_amdgcn_tensor_load_to_lds) && __has_builtin(__builtin_amdgcn_s_wait_tensorcnt)
#define HAVE_TDM 1
#else
#define HAVE_TDM 0
#endif

static __device__ inline v8f wmma_bf16(v16bf a, v16bf b, v8f c) {
  return __builtin_amdgcn_wmma_f32_16x16x32_bf16(false, a, false, b, (short)0, c, false, false);
}

// ---------------- workspace layout (bytes) ----------------
#define OFF_EBF   ((size_t)0)                          // N*D bf16      = 4194304
#define OFF_SQ    ((size_t)4194304)                    // N f32
#define OFF_INVN  ((size_t)4227072)                    // N f32
#define OFF_MEAN  ((size_t)4259840)                    // D f32
#define OFF_SCAL  ((size_t)4260864)                    // small scalars
#define OFF_PDEG  ((size_t)4261120)                    // SPLITS*N f32
#define OFF_PPOT  ((size_t)4523264)
#define OFF_PMX   ((size_t)4785408)
#define OFF_PZ    ((size_t)5047552)
#define OFF_PS1   ((size_t)5309696)
#define OFF_TOPD  ((size_t)5571840)                    // SPLITS*N*6 f32
#define OFF_TOPI  ((size_t)7144704)                    // SPLITS*N*6 i32
#define OFF_NB    ((size_t)8717568)                    // N*5 i32

// LDS layout for k2 (dynamic)
#define SM_ADJ   0            // 256*33*4 = 33792
#define SM_EMBJ  33792        // 32*264*2 = 16896
#define SM_GT    50688        // 256*33*4 = 33792
#define SM_INVNJ 84480        // 128
#define SM_SQJ   84608        // 128
#define SM_TOTAL 84736

// ================= init =================
__global__ void kinit(float* meansum, float* scal) {
  int t = threadIdx.x;
  if (t < DD) meansum[t] = 0.f;
  if (t < 8)  scal[t] = 0.f;
}

// ================= K1: per-row stats + bf16 convert =================
__global__ __launch_bounds__(256) void k1(const float* __restrict__ emb,
                                          __bf16* __restrict__ ebf,
                                          float* __restrict__ sqv,
                                          float* __restrict__ invn) {
  __shared__ float red[256];
  const int i = blockIdx.x, t = threadIdx.x;
  float v = emb[(size_t)i * DD + t];
  ebf[(size_t)i * DD + t] = (__bf16)v;
  red[t] = v * v;
  __syncthreads();
  for (int off = 128; off > 0; off >>= 1) {
    if (t < off) red[t] += red[t + off];
    __syncthreads();
  }
  if (t == 0) {
    float s = red[0];
    sqv[i] = s;
    invn[i] = 1.0f / fmaxf(sqrtf(s), 1e-12f);
  }
}

// ================= K1m: column sums for the global mean =================
__global__ __launch_bounds__(256) void k1m(const float* __restrict__ emb, float* meansum) {
  const int b = blockIdx.x, t = threadIdx.x;
  float s = 0.f;
  for (int r = 0; r < 32; ++r) s += emb[(size_t)(b * 32 + r) * DD + t];
  atomicAdd(&meansum[t], s);
}

// ================= K1b: maxsq, finalize mean, ||mean||^2 =================
__global__ __launch_bounds__(256) void k1b(const float* __restrict__ sqv,
                                           float* __restrict__ mean,
                                           float* __restrict__ scal) {
  __shared__ float red[256];
  const int t = threadIdx.x;
  float m = 0.f;
  for (int k = 0; k < 32; ++k) m = fmaxf(m, sqv[t + 256 * k]);
  red[t] = m;
  __syncthreads();
  for (int off = 128; off > 0; off >>= 1) {
    if (t < off) red[t] = fmaxf(red[t], red[t + off]);
    __syncthreads();
  }
  if (t == 0) scal[0] = red[0];   // max of sq
  __syncthreads();
  float mn = mean[t] * (1.0f / (float)NN);
  mean[t] = mn;
  __syncthreads();
  red[t] = mn * mn;
  __syncthreads();
  for (int off = 128; off > 0; off >>= 1) {
    if (t < off) red[t] += red[t + off];
    __syncthreads();
  }
  if (t == 0) scal[1] = red[0];   // ||mean||^2
}

// ================= K2: fused Gram/adj streaming kernel =================
__global__ __launch_bounds__(256) void k2(const float* __restrict__ adj,
                                          const __bf16* __restrict__ ebf,
                                          const float* __restrict__ sqv,
                                          const float* __restrict__ invn,
                                          float* __restrict__ pdeg, float* __restrict__ ppot,
                                          float* __restrict__ pmx, float* __restrict__ pZ,
                                          float* __restrict__ pS1,
                                          float* __restrict__ ptopd, int* __restrict__ ptopi) {
  extern __shared__ unsigned char smem[];
  float*  adjT  = (float*)(smem + SM_ADJ);
  __bf16* embJ  = (__bf16*)(smem + SM_EMBJ);
  float*  Gt    = (float*)(smem + SM_GT);
  float*  invnJ = (float*)(smem + SM_INVNJ);
  float*  sqJ   = (float*)(smem + SM_SQJ);

  const int tid  = threadIdx.x;
  const int lane = tid & 31;
  const int wv   = tid >> 5;
  const int l16  = lane & 15;
  const int hi   = lane >> 4;
  const int i0   = blockIdx.x * RB;
  const int j0b  = blockIdx.y * JRANGE;

  // ---- preload A fragments (2 row tiles x 8 K-steps) into VGPRs ----
  v16bf A[2][8];
#pragma unroll
  for (int t = 0; t < 2; ++t) {
    const __bf16* rowp = ebf + (size_t)(i0 + (wv * 2 + t) * 16 + l16) * DD;
#pragma unroll
    for (int k = 0; k < 8; ++k) {
      ABu u;
      const __bf16* p = rowp + k * 32 + hi * 8;
      u.q[0] = *(const uint4*)(p);       // K {base..base+7}
      u.q[1] = *(const uint4*)(p + 16);  // K {base+16..base+23}
      A[t][k] = u.v;
    }
  }

  const int irow = i0 + tid;
  const float ci  = invn[irow];
  const float sqi = sqv[irow];
  float deg = 0.f, pot = 0.f;
  float mx = -1e30f, Z = 0.f, S1 = 0.f;
  float bd[6]; int bi[6];
#pragma unroll
  for (int s = 0; s < 6; ++s) { bd[s] = 3.4e38f; bi[s] = -1; }

  for (int ch = 0; ch < CHUNKS; ++ch) {
    const int j0 = j0b + ch * JC;
    __syncthreads();  // previous chunk's consumers done

    // ---- stage embJ (32 rows of bf16, padded stride) ----
#pragma unroll
    for (int k = 0; k < 4; ++k) {
      int l = tid + 256 * k;
      int r = l >> 5, c = l & 31;
      *(uint4*)(embJ + r * EJS + c * 8) =
          *(const uint4*)(ebf + (size_t)(j0 + r) * DD + c * 8);
    }
    if (tid >= 32 && tid < 64) invnJ[tid - 32] = invn[j0 + tid - 32];
    if (tid >= 64 && tid < 96) sqJ[tid - 64] = sqv[j0 + tid - 64];

    // ---- adj tile (256 x 32 f32) -> LDS, padded to stride 33 ----
#if HAVE_TDM
    if (wv == 0) {
      unsigned long long ga =
          (unsigned long long)(const void*)(adj + (size_t)i0 * NN + j0);
      v4u g0 = {0u, 0u, 0u, 0u};
      g0[0] = 1u;                                   // count=1
      g0[1] = (unsigned)SM_ADJ;                     // lds_addr
      g0[2] = (unsigned)ga;
      g0[3] = (unsigned)((ga >> 32) & 0x1FFFFFFull) | (2u << 30);  // type=2
      v8i g1 = {0, 0, 0, 0, 0, 0, 0, 0};
      // data_size=4B(code2), pad_enable, pad_interval=32dw(code4), pad_amount=1dw(0)
      g1[0] = (2 << 16) | (1 << 20) | (4 << 22);
      g1[1] = (int)((NN & 0xFFFF) << 16);           // tensor_dim0 lo16
      g1[2] = (int)((NN >> 16) | ((NN & 0xFFFF) << 16)); // dim0 hi | dim1 lo
      g1[3] = (int)((NN >> 16) | (JC << 16));       // dim1 hi | tile_dim0=32
      g1[4] = RB;                                   // tile_dim1=256, tile_dim2=0
      g1[5] = NN;                                   // tensor_dim0_stride lo32
      g1[6] = 0; g1[7] = 0;
      v4i gz = {0, 0, 0, 0};
      v8i gz8 = {0, 0, 0, 0, 0, 0, 0, 0};
      __builtin_amdgcn_tensor_load_to_lds(g0, g1, gz, gz, gz8, 0);
      __builtin_amdgcn_s_wait_tensorcnt((short)0);
    }
#else
#pragma unroll
    for (int k = 0; k < 8; ++k) {
      int l = tid + 256 * k;
      int r = l >> 3, c4 = (l & 7) * 4;
      float4 v = *(const float4*)(adj + (size_t)(i0 + r) * NN + j0 + c4);
      float* dst = adjT + r * LDW + c4;
      dst[0] = v.x; dst[1] = v.y; dst[2] = v.z; dst[3] = v.w;
    }
#endif
    __syncthreads();

    // ---- WMMA: G tile = E_I (bf16) x E_J^T (bf16), f32 accum ----
#pragma unroll
    for (int c = 0; c < 2; ++c) {
      v8f a0 = {}; v8f a1 = {};
#pragma unroll
      for (int k = 0; k < 8; ++k) {
        ABu bu;
        const __bf16* bp = embJ + (c * 16 + l16) * EJS + k * 32 + hi * 8;
        bu.q[0] = *(const uint4*)(bp);
        bu.q[1] = *(const uint4*)(bp + 16);
        a0 = wmma_bf16(A[0][k], bu.v, a0);
        a1 = wmma_bf16(A[1][k], bu.v, a1);
      }
      {
        int rbase = (wv * 2 + 0) * 16 + hi * 8;
        float* gp = Gt + (size_t)rbase * LDW + c * 16 + l16;
#pragma unroll
        for (int v = 0; v < 8; ++v) gp[v * LDW] = a0[v];
      }
      {
        int rbase = (wv * 2 + 1) * 16 + hi * 8;
        float* gp = Gt + (size_t)rbase * LDW + c * 16 + l16;
#pragma unroll
        for (int v = 0; v < 8; ++v) gp[v * LDW] = a1[v];
      }
    }
    __syncthreads();

    // ---- per-row streaming reduction: thread owns row `tid` ----
    const float* gr = Gt + (size_t)tid * LDW;
    const float* ar = adjT + (size_t)tid * LDW;
#pragma unroll 4
    for (int jj = 0; jj < JC; ++jj) {
      float g = gr[jj];
      float a = ar[jj];
      deg += a;
      pot = fmaf(a, g, pot);
      // masked similarity and online softmax stats
      float x = g * ci * (invnJ[jj] * a);
      if (x > mx) {
        float d = mx - x;
        float e = __expf(d);
        S1 = e * fmaf(d, Z, S1);
        Z = fmaf(Z, e, 1.0f);
        mx = x;
      } else {
        float d = x - mx;
        float e = __expf(d);
        Z += e;
        S1 = fmaf(e, d, S1);
      }
      // streaming top-6 smallest squared distances
      float d2 = sqi + sqJ[jj] - 2.0f * g;
      if (d2 < bd[5]) {
        float v = d2; int ix = j0 + jj;
#pragma unroll
        for (int s = 0; s < 6; ++s) {
          if (v < bd[s]) {
            float tv = bd[s]; int ti = bi[s];
            bd[s] = v; bi[s] = ix; v = tv; ix = ti;
          }
        }
      }
    }
  }

  // ---- write per-split partials ----
  const size_t o = (size_t)blockIdx.y * NN + irow;
  pdeg[o] = deg; ppot[o] = pot; pmx[o] = mx; pZ[o] = Z; pS1[o] = S1;
#pragma unroll
  for (int s = 0; s < 6; ++s) { ptopd[o * 6 + s] = bd[s]; ptopi[o * 6 + s] = bi[s]; }
}

// ================= K2b: merge splits, finalize 3 outputs =================
__global__ __launch_bounds__(256) void k2b(const float* __restrict__ pdeg,
                                           const float* __restrict__ ppot,
                                           const float* __restrict__ pmx,
                                           const float* __restrict__ pZ,
                                           const float* __restrict__ pS1,
                                           const float* __restrict__ ptopd,
                                           const int* __restrict__ ptopi,
                                           const float* __restrict__ sqv,
                                           const float* __restrict__ scal,
                                           int* __restrict__ nbidx,
                                           float* __restrict__ out) {
  const int i = blockIdx.x * 256 + threadIdx.x;
  float deg = 0.f, pot = 0.f;
  float M = -1e30f, Z = 0.f, S1 = 0.f;
  float bd[6]; int bi[6];
#pragma unroll
  for (int s = 0; s < 6; ++s) { bd[s] = 3.4e38f; bi[s] = -1; }

  for (int s = 0; s < SPLITS; ++s) {
    const size_t o = (size_t)s * NN + i;
    deg += pdeg[o];
    pot += ppot[o];
    float m = pmx[o], z = pZ[o], s1 = pS1[o];
    if (m > M) {
      float d = M - m;
      float e = __expf(d);
      S1 = e * fmaf(d, Z, S1) + s1;
      Z = fmaf(Z, e, z);
      M = m;
    } else {
      float d = m - M;
      float e = __expf(d);
      S1 = fmaf(e, fmaf(d, z, s1), S1);
      Z = fmaf(e, z, Z);
    }
#pragma unroll
    for (int k = 0; k < 6; ++k) {
      float v = ptopd[o * 6 + k];
      int ix = ptopi[o * 6 + k];
      if (v < bd[5]) {
#pragma unroll
        for (int q = 0; q < 6; ++q) {
          if (v < bd[q]) {
            float tv = bd[q]; int ti = bi[q];
            bd[q] = v; bi[q] = ix; v = tv; ix = ti;
          }
        }
      }
    }
  }

  const float sqi = sqv[i];
  float ent = (Z > 0.f) ? (logf(Z) - S1 / Z) : 0.f;
  out[i] = sqi - pot / (deg + EPS);                       // total_energy
  out[NN + i] = (deg > 0.f) ? ent : 0.f;                  // local_entropy
  float dc = deg / ((float)(NN - 1) + EPS);
  float ec = sqrtf(sqi) / (sqrtf(scal[0]) + EPS);
  out[3 * (size_t)NN + i] = 0.5f * dc + 0.5f * ec;        // structural_centrality

  // neighbor list: drop self, keep 5 nearest
  int nb[5]; int cnt = 0;
#pragma unroll
  for (int k = 0; k < 6; ++k) {
    if (bi[k] != i && cnt < 5) { nb[cnt] = bi[k]; ++cnt; }
  }
  for (int k = cnt; k < 5; ++k) nb[k] = (bi[5] >= 0) ? bi[5] : i;
#pragma unroll
  for (int k = 0; k < 5; ++k) nbidx[(size_t)i * 5 + k] = nb[k];
}

// ================= K3: equilibrium deviation =================
__global__ __launch_bounds__(256) void k3(const float* __restrict__ emb,
                                          const float* __restrict__ mean,
                                          const float* __restrict__ sqv,
                                          const float* __restrict__ scal,
                                          const int* __restrict__ nbidx,
                                          float* __restrict__ out) {
  const int w = threadIdx.x >> 5;
  const int lane = threadIdx.x & 31;
  const int i = blockIdx.x * 8 + w;
  int nb[5];
#pragma unroll
  for (int k = 0; k < 5; ++k) nb[k] = nbidx[(size_t)i * 5 + k];
  float accld = 0.f, accdot = 0.f;
#pragma unroll
  for (int k = 0; k < 8; ++k) {
    int d = lane + 32 * k;
    float e = emb[(size_t)i * DD + d];
    float s = 0.f;
#pragma unroll
    for (int q = 0; q < 5; ++q) s += emb[(size_t)nb[q] * DD + d];
    float lm = s * 0.2f;
    float diff = e - lm;
    accld = fmaf(diff, diff, accld);
    accdot = fmaf(e, mean[d], accdot);
  }
  for (int off = 16; off > 0; off >>= 1) {
    accld += __shfl_xor(accld, off, 32);
    accdot += __shfl_xor(accdot, off, 32);
  }
  if (lane == 0) {
    float gd2 = sqv[i] - 2.f * accdot + scal[1];
    float gd = sqrtf(fmaxf(gd2, 0.f));
    float ld = sqrtf(fmaxf(accld, 0.f));
    out[2 * (size_t)NN + i] = 0.6f * gd + 0.4f * ld;
  }
}

// ================= host launcher =================
extern "C" void kernel_launch(void* const* d_in, const int* in_sizes, int n_in,
                              void* d_out, int out_size, void* d_ws, size_t ws_size,
                              hipStream_t stream) {
  (void)in_sizes; (void)n_in; (void)out_size; (void)ws_size;
  const float* emb = (const float*)d_in[0];
  const float* adj = (const float*)d_in[1];
  float* out = (float*)d_out;
  char* ws = (char*)d_ws;

  __bf16* ebf   = (__bf16*)(ws + OFF_EBF);
  float* sqv    = (float*)(ws + OFF_SQ);
  float* invn   = (float*)(ws + OFF_INVN);
  float* mean   = (float*)(ws + OFF_MEAN);
  float* scal   = (float*)(ws + OFF_SCAL);
  float* pdeg   = (float*)(ws + OFF_PDEG);
  float* ppot   = (float*)(ws + OFF_PPOT);
  float* pmx    = (float*)(ws + OFF_PMX);
  float* pZ     = (float*)(ws + OFF_PZ);
  float* pS1    = (float*)(ws + OFF_PS1);
  float* ptopd  = (float*)(ws + OFF_TOPD);
  int*   ptopi  = (int*)(ws + OFF_TOPI);
  int*   nbidx  = (int*)(ws + OFF_NB);

  kinit<<<1, 256, 0, stream>>>(mean, scal);
  k1<<<NN, 256, 0, stream>>>(emb, ebf, sqv, invn);
  k1m<<<NN / 32, 256, 0, stream>>>(emb, mean);
  k1b<<<1, 256, 0, stream>>>(sqv, mean, scal);
  k2<<<dim3(NN / RB, SPLITS), 256, SM_TOTAL, stream>>>(adj, ebf, sqv, invn,
                                                       pdeg, ppot, pmx, pZ, pS1,
                                                       ptopd, ptopi);
  k2b<<<NN / 256, 256, 0, stream>>>(pdeg, ppot, pmx, pZ, pS1, ptopd, ptopi,
                                    sqv, scal, nbidx, out);
  k3<<<NN / 8, 256, 0, stream>>>(emb, mean, sqv, scal, nbidx, out);
}
